// DETRAttention_1812476199681
// MI455X (gfx1250) — compile-verified, matrix operand order checked by reading, not run on hardware
//
#include <hip/hip_runtime.h>

// Problem constants (fixed by the reference)
#define B_ 4
#define T_ 2048
#define E_ 1024
#define H_ 16
#define D_ 64
#define MB_ (B_ * T_)   // 8192 rows for the projection GEMMs

typedef __attribute__((ext_vector_type(16))) __bf16 v16bf;
typedef __attribute__((ext_vector_type(8)))  __bf16 v8bf;
typedef __attribute__((ext_vector_type(8)))  float  v8f;
typedef __attribute__((ext_vector_type(4)))  int    v4i;

#if defined(__has_builtin)
#  if __has_builtin(__builtin_amdgcn_global_load_async_to_lds_b128) && \
      __has_builtin(__builtin_amdgcn_s_wait_asynccnt)
#    define HAVE_ASYNC_LDS 1
#  endif
#endif

// fp32 -> bf16 round-to-nearest-even, stored as raw u16
__device__ __forceinline__ unsigned short f2bf(float f) {
    unsigned int u = __float_as_uint(f);
    u += 0x7FFFu + ((u >> 16) & 1u);
    return (unsigned short)(u >> 16);
}

// Load one 16x32 bf16 WMMA fragment (A layout; B layout is identical with
// lane = N column when the source matrix is row-major [N, K]).
// ISA 7.12.2: lane l<16 -> row l, kb=0; lane l>=16 -> row l-16, kb=8;
// each lane holds K = kb..kb+7 and kb+16..kb+23.
__device__ __forceinline__ v16bf load_frag(const unsigned short* base, int stride, int lane) {
    const unsigned short* p = base + (lane & 15) * stride + ((lane >> 4) << 3);
    v8bf lo = *(const v8bf*)(p);
    v8bf hi = *(const v8bf*)(p + 16);
    return __builtin_shufflevector(lo, hi, 0,1,2,3,4,5,6,7,8,9,10,11,12,13,14,15);
}

__device__ __forceinline__ float redmax16(float x) {
    #pragma unroll
    for (int off = 8; off >= 1; off >>= 1) x = fmaxf(x, __shfl_xor(x, off, 16));
    return x;
}
__device__ __forceinline__ float redsum16(float x) {
    #pragma unroll
    for (int off = 8; off >= 1; off >>= 1) x += __shfl_xor(x, off, 16);
    return x;
}

// ---------------------------------------------------------------- elementwise
__global__ void prep_kernel(const float* __restrict__ hs, const float* __restrict__ oq,
                            unsigned short* __restrict__ hqb, unsigned short* __restrict__ hsb,
                            int n) {
    int i = blockIdx.x * 256 + threadIdx.x;
    if (i < n) {
        float a = hs[i];
        hqb[i] = f2bf(a + oq[i]);
        hsb[i] = f2bf(a);
    }
}

__global__ void cvt_kernel(const float* __restrict__ src, unsigned short* __restrict__ dst, int n) {
    int i = blockIdx.x * 256 + threadIdx.x;
    if (i < n) dst[i] = f2bf(src[i]);
}

// ---------------------------------------------------------------- GEMM
// C[m,n] = (sum_k A[m,k] * W[n,k] + bias[n]) * alpha
// mode 0: bf16 out, row-major [M, E]                 (Q, K projections)
// mode 1: bf16 out, per-head transposed Vt[b,h,d,t]  (V projection)
// mode 2: fp32 out, row-major [M, E]                 (output projection)
//
// The 64x32 W slab is identical for all 4 waves of a block: it is staged in a
// double-buffered LDS tile via GLOBAL_LOAD_ASYNC_TO_LDS_B128 (ASYNCcnt DMA),
// cutting W global traffic 4x. A fragments (distinct rows per wave) use a
// global->register ping-pong so their loads overlap the WMMAs.
__global__ __launch_bounds__(128) void gemm_bf16_wmma(
    const unsigned short* __restrict__ A,
    const unsigned short* __restrict__ W,
    const float* __restrict__ bias,
    unsigned short* __restrict__ outb,
    float* __restrict__ outf,
    float alpha, int mode) {
    const int lane = threadIdx.x & 31;
    const int wave = threadIdx.x >> 5;
    const int m0 = blockIdx.y * 128 + wave * 32;
    const int n0 = blockIdx.x * 64;

    const unsigned short* Abase = A + m0 * E_;
    const unsigned short* Wbase = W + n0 * E_;
    const unsigned short* Arow = Abase + (lane & 15) * E_;   // prefetch addressing

    v8f acc[2][4];
    #pragma unroll
    for (int i = 0; i < 2; i++)
        #pragma unroll
        for (int j = 0; j < 4; j++)
            acc[i][j] = (v8f){0.f,0.f,0.f,0.f,0.f,0.f,0.f,0.f};

#ifdef HAVE_ASYNC_LDS
    // [2 buffers][64 rows][32 k] bf16 = 8 KB
    __shared__ __align__(16) unsigned short ldsW[2][64 * 32];
    const int tid = threadIdx.x;
    const int wrow = tid >> 2;          // rows wrow and wrow+32
    const int wkc  = (tid & 3) * 8;     // 16B k-chunk within the 32-k slab

    // Each thread DMAs two 16B chunks of the W slab at k-offset (kk) into buf.
    #define ISSUE_W(buf, kk) do {                                                         \
        __builtin_amdgcn_global_load_async_to_lds_b128(                                   \
            (__attribute__((address_space(1))) v4i*)(Wbase + wrow * E_ + (kk) + wkc),     \
            (__attribute__((address_space(3))) v4i*)(&ldsW[buf][wrow * 32 + wkc]), 0, 0); \
        __builtin_amdgcn_global_load_async_to_lds_b128(                                   \
            (__attribute__((address_space(1))) v4i*)(Wbase + (wrow + 32) * E_ + (kk) + wkc), \
            (__attribute__((address_space(3))) v4i*)(&ldsW[buf][(wrow + 32) * 32 + wkc]), 0, 0); \
    } while (0)

    ISSUE_W(0, 0);
    ISSUE_W(1, 32);

    v16bf aA[2], aB[2];
    #pragma unroll
    for (int i = 0; i < 2; i++) aA[i] = load_frag(Abase + i * 16 * E_, E_, lane);

    for (int k0 = 0; k0 < E_; k0 += 64) {
        __builtin_prefetch(Arow + k0 + 256, 0, 1);
        __builtin_prefetch(Arow + 16 * E_ + k0 + 256, 0, 1);

        // ---- half 1: consume ldsW[0] (k0) with aA ----
        #pragma unroll
        for (int i = 0; i < 2; i++) aB[i] = load_frag(Abase + i * 16 * E_ + k0 + 32, E_, lane);

        __builtin_amdgcn_s_wait_asynccnt(2);   // own buf0 chunks landed
        __syncthreads();                       // all waves' chunks landed
        {
            v16bf bw[4];
            #pragma unroll
            for (int j = 0; j < 4; j++) bw[j] = load_frag(&ldsW[0][j * 16 * 32], 32, lane);
            #pragma unroll
            for (int i = 0; i < 2; i++)
                #pragma unroll
                for (int j = 0; j < 4; j++)
                    acc[i][j] = __builtin_amdgcn_wmma_f32_16x16x32_bf16(
                        false, aA[i], false, bw[j], (short)0, acc[i][j], false, false);
        }
        __syncthreads();                       // all waves done reading buf0
        ISSUE_W(0, (k0 + 64) & (E_ - 1));      // refill (wraps harmlessly at end)

        // ---- half 2: consume ldsW[1] (k0+32) with aB ----
        const int k2 = (k0 + 64) & (E_ - 1);
        #pragma unroll
        for (int i = 0; i < 2; i++) aA[i] = load_frag(Abase + i * 16 * E_ + k2, E_, lane);

        __builtin_amdgcn_s_wait_asynccnt(2);
        __syncthreads();
        {
            v16bf bw[4];
            #pragma unroll
            for (int j = 0; j < 4; j++) bw[j] = load_frag(&ldsW[1][j * 16 * 32], 32, lane);
            #pragma unroll
            for (int i = 0; i < 2; i++)
                #pragma unroll
                for (int j = 0; j < 4; j++)
                    acc[i][j] = __builtin_amdgcn_wmma_f32_16x16x32_bf16(
                        false, aB[i], false, bw[j], (short)0, acc[i][j], false, false);
        }
        __syncthreads();
        ISSUE_W(1, (k0 + 96) & (E_ - 1));
    }
    #undef ISSUE_W
#else
    // Fallback: all-register 2x-unrolled ping-pong (W per wave from global).
    const unsigned short* Wrow = Wbase + (lane & 15) * E_;
    v16bf aA[2], bA[4];
    #pragma unroll
    for (int i = 0; i < 2; i++) aA[i] = load_frag(Abase + i * 16 * E_, E_, lane);
    #pragma unroll
    for (int j = 0; j < 4; j++) bA[j] = load_frag(Wbase + j * 16 * E_, E_, lane);

    for (int k0 = 0; k0 < E_; k0 += 64) {
        __builtin_prefetch(Arow + k0 + 256, 0, 1);
        __builtin_prefetch(Arow + 16 * E_ + k0 + 256, 0, 1);
        #pragma unroll
        for (int j = 0; j < 4; j++) __builtin_prefetch(Wrow + j * 16 * E_ + k0 + 256, 0, 1);

        v16bf aB[2], bB[4];
        #pragma unroll
        for (int i = 0; i < 2; i++) aB[i] = load_frag(Abase + i * 16 * E_ + k0 + 32, E_, lane);
        #pragma unroll
        for (int j = 0; j < 4; j++) bB[j] = load_frag(Wbase + j * 16 * E_ + k0 + 32, E_, lane);

        #pragma unroll
        for (int i = 0; i < 2; i++)
            #pragma unroll
            for (int j = 0; j < 4; j++)
                acc[i][j] = __builtin_amdgcn_wmma_f32_16x16x32_bf16(
                    false, aA[i], false, bA[j], (short)0, acc[i][j], false, false);

        const int k2 = (k0 + 64) & (E_ - 1);
        #pragma unroll
        for (int i = 0; i < 2; i++) aA[i] = load_frag(Abase + i * 16 * E_ + k2, E_, lane);
        #pragma unroll
        for (int j = 0; j < 4; j++) bA[j] = load_frag(Wbase + j * 16 * E_ + k2, E_, lane);

        #pragma unroll
        for (int i = 0; i < 2; i++)
            #pragma unroll
            for (int j = 0; j < 4; j++)
                acc[i][j] = __builtin_amdgcn_wmma_f32_16x16x32_bf16(
                    false, aB[i], false, bB[j], (short)0, acc[i][j], false, false);
    }
#endif

    const int rowoff = (lane >> 4) << 3;
    #pragma unroll
    for (int i = 0; i < 2; i++) {
        #pragma unroll
        for (int j = 0; j < 4; j++) {
            const int n = n0 + j * 16 + (lane & 15);
            const float bv = bias[n];
            #pragma unroll
            for (int r = 0; r < 8; r++) {
                const int m = m0 + i * 16 + rowoff + r;
                const float val = (acc[i][j][r] + bv) * alpha;
                if (mode == 0) {
                    outb[m * E_ + n] = f2bf(val);
                } else if (mode == 1) {
                    const int t = m & (T_ - 1);
                    const int bb = m >> 11;               // m / T_
                    const int h = n >> 6;                 // n / D_
                    const int d = n & (D_ - 1);
                    outb[((bb * H_ + h) * D_ + d) * T_ + t] = f2bf(val);
                } else {
                    outf[m * E_ + n] = val;
                }
            }
        }
    }
}

// ---------------------------------------------------------------- attention
// One wave per (b, h, 16-row Q tile). Online softmax over s in steps of 32.
// All s-step loads issued up front; next s-step warmed via global_prefetch_b8.
__global__ __launch_bounds__(32) void attn_wmma(
    const unsigned short* __restrict__ Q,   // [B,T,E] bf16 (scale pre-applied)
    const unsigned short* __restrict__ K,   // [B,T,E] bf16
    const unsigned short* __restrict__ Vt,  // [B,H,D,T] bf16
    unsigned short* __restrict__ AO) {      // [B,T,E] bf16
    __shared__ __align__(16) unsigned short ldsP[16 * 32];

    const int lane = threadIdx.x & 31;
    const int wg = blockIdx.x;
    const int t0 = (wg & (T_ / 16 - 1)) * 16;
    const int bh = wg >> 7;                 // wg / (T_/16)
    const int h = bh & (H_ - 1);
    const int b = bh >> 4;

    const unsigned short* Kbase = K + b * T_ * E_ + h * D_;
    const unsigned short* Vbase = Vt + (b * H_ + h) * D_ * T_;

    v16bf qf[2];
    #pragma unroll
    for (int kk = 0; kk < 2; kk++)
        qf[kk] = load_frag(Q + (b * T_ + t0) * E_ + h * D_ + kk * 32, E_, lane);

    v8f o[4];
    #pragma unroll
    for (int j = 0; j < 4; j++) o[j] = (v8f){0.f,0.f,0.f,0.f,0.f,0.f,0.f,0.f};
    float rm[8], rl[8];
    #pragma unroll
    for (int r = 0; r < 8; r++) { rm[r] = -1e30f; rl[r] = 0.f; }

    const int rowoff = (lane >> 4) << 3;

    for (int s0 = 0; s0 < T_; s0 += 32) {
        // Issue ALL loads for this s-step up front: V (consumed last) first,
        // then K fragments for both 16-row sub-tiles.
        v16bf vf[4];
        #pragma unroll
        for (int j = 0; j < 4; j++)
            vf[j] = load_frag(Vbase + (j * 16) * T_ + s0, T_, lane);
        v16bf kf1[2], kf2[2];
        #pragma unroll
        for (int kk = 0; kk < 2; kk++) {
            kf1[kk] = load_frag(Kbase + s0 * E_ + kk * 32, E_, lane);
            kf2[kk] = load_frag(Kbase + (s0 + 16) * E_ + kk * 32, E_, lane);
        }

        // Warm next s-step: 32 K rows (one per lane) and 64 V rows (two per lane)
        const int sn = (s0 + 32) & (T_ - 1);
        __builtin_prefetch(Kbase + (sn + lane) * E_, 0, 1);
        __builtin_prefetch(Vbase + lane * T_ + sn, 0, 1);
        __builtin_prefetch(Vbase + (lane + 32) * T_ + sn, 0, 1);

        v8f s1 = (v8f){0.f,0.f,0.f,0.f,0.f,0.f,0.f,0.f};
        v8f s2 = (v8f){0.f,0.f,0.f,0.f,0.f,0.f,0.f,0.f};
        #pragma unroll
        for (int kk = 0; kk < 2; kk++) {
            s1 = __builtin_amdgcn_wmma_f32_16x16x32_bf16(false, qf[kk], false, kf1[kk], (short)0, s1, false, false);
            s2 = __builtin_amdgcn_wmma_f32_16x16x32_bf16(false, qf[kk], false, kf2[kk], (short)0, s2, false, false);
        }

        // Online softmax: row r of C-fragment VGPR r spans 16 lanes of this half.
        #pragma unroll
        for (int r = 0; r < 8; r++) {
            float mx = redmax16(fmaxf(s1[r], s2[r]));
            float mnew = fmaxf(rm[r], mx);
            float al = __expf(rm[r] - mnew);
            float p1 = __expf(s1[r] - mnew);
            float p2 = __expf(s2[r] - mnew);
            float rs = redsum16(p1 + p2);
            rl[r] = rl[r] * al + rs;
            rm[r] = mnew;
            #pragma unroll
            for (int j = 0; j < 4; j++) o[j][r] *= al;
            const int row = rowoff + r;
            ldsP[row * 32 + (lane & 15)]      = f2bf(p1);
            ldsP[row * 32 + 16 + (lane & 15)] = f2bf(p2);
        }
        __syncthreads();
        v16bf pa = load_frag(ldsP, 32, lane);   // 16x32 A fragment of P
        #pragma unroll
        for (int j = 0; j < 4; j++) {
            o[j] = __builtin_amdgcn_wmma_f32_16x16x32_bf16(false, pa, false, vf[j], (short)0, o[j], false, false);
        }
        __syncthreads();
    }

    #pragma unroll
    for (int r = 0; r < 8; r++) {
        const float inv = 1.0f / rl[r];
        const int t = t0 + rowoff + r;
        #pragma unroll
        for (int j = 0; j < 4; j++)
            AO[(b * T_ + t) * E_ + h * D_ + j * 16 + (lane & 15)] = f2bf(o[j][r] * inv);
    }
}

// ---------------------------------------------------------------- launcher
extern "C" void kernel_launch(void* const* d_in, const int* in_sizes, int n_in,
                              void* d_out, int out_size, void* d_ws, size_t ws_size,
                              hipStream_t stream) {
    (void)in_sizes; (void)n_in; (void)out_size; (void)ws_size;
    const float* hs = (const float*)d_in[0];
    const float* oq = (const float*)d_in[1];
    const float* Wq = (const float*)d_in[2];
    const float* bq = (const float*)d_in[3];
    const float* Wk = (const float*)d_in[4];
    const float* bk = (const float*)d_in[5];
    const float* Wv = (const float*)d_in[6];
    const float* bv = (const float*)d_in[7];
    const float* Wo = (const float*)d_in[8];
    const float* bo = (const float*)d_in[9];

    const size_t BTE = (size_t)B_ * T_ * E_;   // 8,388,608
    const size_t EE  = (size_t)E_ * E_;        // 1,048,576

    unsigned short* ws  = (unsigned short*)d_ws;
    unsigned short* hqb = ws;                  // [B,T,E] bf16  (hs + oq)
    unsigned short* hsb = hqb + BTE;           // [B,T,E] bf16  (hs)
    unsigned short* Wqb = hsb + BTE;
    unsigned short* Wkb = Wqb + EE;
    unsigned short* Wvb = Wkb + EE;
    unsigned short* Wob = Wvb + EE;
    unsigned short* Qb  = Wob + EE;            // [B,T,E]
    unsigned short* Kb  = Qb + BTE;            // [B,T,E]
    unsigned short* Vtb = Kb + BTE;            // [B,H,D,T]
    unsigned short* AOb = hqb;                 // reuse: hq dead after Q/K GEMMs

    prep_kernel<<<(int)(BTE / 256), 256, 0, stream>>>(hs, oq, hqb, hsb, (int)BTE);
    cvt_kernel<<<(int)(EE / 256), 256, 0, stream>>>(Wq, Wqb, (int)EE);
    cvt_kernel<<<(int)(EE / 256), 256, 0, stream>>>(Wk, Wkb, (int)EE);
    cvt_kernel<<<(int)(EE / 256), 256, 0, stream>>>(Wv, Wvb, (int)EE);
    cvt_kernel<<<(int)(EE / 256), 256, 0, stream>>>(Wo, Wob, (int)EE);

    dim3 gg(E_ / 64, MB_ / 128);  // (16, 64), 128 threads = 4 waves, 128x64 C tile
    const float scale = 0.125f;   // D^-0.5 = 1/8, folded into Q
    gemm_bf16_wmma<<<gg, 128, 0, stream>>>(hqb, Wqb, bq, Qb,  nullptr, scale, 0);
    gemm_bf16_wmma<<<gg, 128, 0, stream>>>(hqb, Wkb, bk, Kb,  nullptr, 1.0f,  0);
    gemm_bf16_wmma<<<gg, 128, 0, stream>>>(hsb, Wvb, bv, Vtb, nullptr, 1.0f,  1);

    attn_wmma<<<B_ * H_ * (T_ / 16), 32, 0, stream>>>(Qb, Kb, Vtb, AOb);

    gemm_bf16_wmma<<<gg, 128, 0, stream>>>(AOb, Wob, bo, nullptr, (float*)d_out, 1.0f, 2);
}
